// BinaryLinear_1623497638680
// MI455X (gfx1250) — compile-verified, matrix operand order checked by reading, not run on hardware
//
#include <hip/hip_runtime.h>
#include <stdint.h>

// Problem dims (fixed by reference)
#define M_DIM 16384
#define N_DIM 2048
#define K_DIM 2048

// GEMM tiling: block 128x256, K-stage 32, 8 waves of 32, wave tile 64x64.
#define BM 128
#define BN 256
#define BK 32
#define SH_STRIDE_E 40                    // padded bf16 elems per LDS row (80B, conflict-free)
#define A_STAGE_E   (BM * SH_STRIDE_E)    // 5120 elems
#define B_STAGE_E   (BN * SH_STRIDE_E)    // 10240 elems
#define STAGE_E     (A_STAGE_E + B_STAGE_E) // 15360 elems = 30720 B per stage
#define LDS_BYTES   (2 * STAGE_E * 2)     // double buffered = 61440 bytes (< 64KB)

typedef __attribute__((ext_vector_type(16))) __bf16          bf16x16;
typedef __attribute__((ext_vector_type(8)))  unsigned short  u16x8;
typedef __attribute__((ext_vector_type(8)))  float           v8f;

union FragU {
    u16x8   h[2];
    bf16x16 v;
};

__device__ __forceinline__ unsigned short f32_to_bf16_rne(float f) {
    unsigned u = __float_as_uint(f);
    unsigned r = 0x7FFFu + ((u >> 16) & 1u);
    return (unsigned short)((u + r) >> 16);
}

__device__ __forceinline__ void wait_async0() {
#if __has_builtin(__builtin_amdgcn_s_wait_asynccnt)
    __builtin_amdgcn_s_wait_asynccnt(0);
#else
    asm volatile("s_wait_asynccnt 0" ::: "memory");
#endif
}

// Async global->LDS DMA (ASYNCcnt-tracked). vdst = LDS byte offset, vaddr = 64-bit global.
#define ASYNC_LD_B128(ldsoff, gptr)                                             \
    asm volatile("global_load_async_to_lds_b128 %0, %1, off"                    \
                 :: "v"((unsigned)(ldsoff)),                                    \
                    "v"((unsigned long long)(uintptr_t)(gptr)) : "memory")

// ---------------- elementwise preprocessing ----------------

__global__ void cvt_x_bf16(const float* __restrict__ x,
                           unsigned short* __restrict__ xb, int n4) {
    int i = blockIdx.x * blockDim.x + threadIdx.x;
    if (i >= n4) return;
    float4 v = ((const float4*)x)[i];
    ushort4 o;
    o.x = f32_to_bf16_rne(v.x);
    o.y = f32_to_bf16_rne(v.y);
    o.z = f32_to_bf16_rne(v.z);
    o.w = f32_to_bf16_rne(v.w);
    ((ushort4*)xb)[i] = o;
}

__global__ void cvt_w_sign(const float* __restrict__ w,
                           unsigned short* __restrict__ wb, int n4) {
    int i = blockIdx.x * blockDim.x + threadIdx.x;
    if (i >= n4) return;
    float4 v = ((const float4*)w)[i];
    ushort4 o;
    o.x = v.x > 0.f ? 0x3F80u : (v.x < 0.f ? 0xBF80u : 0u);  // +1 / -1 / 0 in bf16
    o.y = v.y > 0.f ? 0x3F80u : (v.y < 0.f ? 0xBF80u : 0u);
    o.z = v.z > 0.f ? 0x3F80u : (v.z < 0.f ? 0xBF80u : 0u);
    o.w = v.w > 0.f ? 0x3F80u : (v.w < 0.f ? 0xBF80u : 0u);
    ((ushort4*)wb)[i] = o;
}

// ---------------- main WMMA GEMM ----------------
// out[m,n] = alpha * sum_k xbf[m,k] * signW[n,k]

__global__ __launch_bounds__(256) void bgemm_wmma(
    const unsigned short* __restrict__ A,   // x  bf16 [M][K]
    const unsigned short* __restrict__ Bw,  // sign(W) bf16 [N][K]
    const float* __restrict__ alphap,
    float* __restrict__ C)                  // [M][N] f32
{
    extern __shared__ __align__(16) unsigned short smem[];

    const int tid    = threadIdx.x;
    const int lane   = tid & 31;
    const int wave   = tid >> 5;
    const int wm     = wave >> 2;          // 0..1 : 64-row wave tile
    const int wn     = wave & 3;           // 0..3 : 64-col wave tile
    const int lane15 = lane & 15;
    const int hi     = lane >> 4;          // K-half select per WMMA layout

    const int row0 = blockIdx.y * BM;
    const int col0 = blockIdx.x * BN;

    // LDS byte base (low 32 bits of flat shared address = wave LDS offset)
    const unsigned sbase = (unsigned)(unsigned long long)(const void*)smem;

    // Global->LDS staging: chunks of 8 bf16; A tile 128x32 = 512 chunks (2/thread),
    // B tile 256x32 = 1024 chunks (4/thread). Same k-column for all of a thread's chunks.
    const unsigned rr = tid >> 2;                  // base row 0..63
    const unsigned cc = (tid & 3) * 8;             // k-col within stage (elems)

    const char* gA[2];
    const char* gB[4];
    unsigned    ldsA[2], ldsB[4];
#pragma unroll
    for (int i = 0; i < 2; ++i) {
        unsigned r = rr + 64u * i;
        gA[i]   = (const char*)(A + (size_t)(row0 + r) * K_DIM + cc);
        ldsA[i] = sbase + (r * SH_STRIDE_E + cc) * 2;
    }
#pragma unroll
    for (int i = 0; i < 4; ++i) {
        unsigned r = rr + 64u * i;
        gB[i]   = (const char*)(Bw + (size_t)(col0 + r) * K_DIM + cc);
        ldsB[i] = sbase + (A_STAGE_E + r * SH_STRIDE_E + cc) * 2;
    }

    // Fragment LDS element offsets (per documented 16-bit WMMA A/B layouts):
    // lane row = lane&15; lanes 16..31 shifted by 8 in K; second chunk at +16 K.
    unsigned aoffE[4], boffE[4];
#pragma unroll
    for (int ti = 0; ti < 4; ++ti)
        aoffE[ti] = (unsigned)((wm * 64 + ti * 16 + lane15) * SH_STRIDE_E + hi * 8);
#pragma unroll
    for (int tj = 0; tj < 4; ++tj)
        boffE[tj] = (unsigned)(A_STAGE_E + (wn * 64 + tj * 16 + lane15) * SH_STRIDE_E + hi * 8);

    v8f acc[4][4];
#pragma unroll
    for (int i = 0; i < 4; ++i)
#pragma unroll
        for (int j = 0; j < 4; ++j)
            acc[i][j] = 0.0f;

    auto issue = [&](int buf, int kByte) {
        const unsigned so = (unsigned)(buf * STAGE_E * 2);
#pragma unroll
        for (int i = 0; i < 2; ++i)
            ASYNC_LD_B128(ldsA[i] + so, gA[i] + kByte);
#pragma unroll
        for (int i = 0; i < 4; ++i)
            ASYNC_LD_B128(ldsB[i] + so, gB[i] + kByte);
    };

    auto compute = [&](int buf) {
        const unsigned sb = (unsigned)(buf * STAGE_E);
        FragU b[4];
#pragma unroll
        for (int tj = 0; tj < 4; ++tj) {
            b[tj].h[0] = *(const u16x8*)(&smem[sb + boffE[tj]]);
            b[tj].h[1] = *(const u16x8*)(&smem[sb + boffE[tj] + 16]);
        }
        // Register-double-buffered A fragments: load a[ti+1] before consuming a[ti],
        // so each ds_load overlaps the previous group's 4 WMMAs (wait dscnt<=2, not 0).
        FragU a[2];
        a[0].h[0] = *(const u16x8*)(&smem[sb + aoffE[0]]);
        a[0].h[1] = *(const u16x8*)(&smem[sb + aoffE[0] + 16]);
#pragma unroll
        for (int ti = 0; ti < 4; ++ti) {
            if (ti < 3) {
                a[(ti + 1) & 1].h[0] = *(const u16x8*)(&smem[sb + aoffE[ti + 1]]);
                a[(ti + 1) & 1].h[1] = *(const u16x8*)(&smem[sb + aoffE[ti + 1] + 16]);
            }
#pragma unroll
            for (int tj = 0; tj < 4; ++tj)
                acc[ti][tj] = __builtin_amdgcn_wmma_f32_16x16x32_bf16(
                    false, a[ti & 1].v, false, b[tj].v,
                    (short)0, acc[ti][tj], false, false);
        }
    };

    // Software pipeline: prefetch stage 0, then overlap stage k+1 DMA with stage k math.
    issue(0, 0);
    wait_async0();
    __syncthreads();

    int buf = 0;
    for (int k0 = 0; k0 < K_DIM; k0 += BK) {
        if (k0 + BK < K_DIM) issue(buf ^ 1, (k0 + BK) * 2);
        compute(buf);
        wait_async0();
        __syncthreads();
        buf ^= 1;
    }

    // Epilogue: C layout = lane n + VGPR r (M = r, lanes>=16 -> M+8). Scale by alpha.
    const float alpha = alphap[0];
#pragma unroll
    for (int ti = 0; ti < 4; ++ti) {
#pragma unroll
        for (int tj = 0; tj < 4; ++tj) {
            const int cn = col0 + wn * 64 + tj * 16 + lane15;
            const int rm = row0 + wm * 64 + ti * 16 + hi * 8;
            float* cp = C + (size_t)rm * N_DIM + cn;
#pragma unroll
            for (int r = 0; r < 8; ++r)
                cp[(size_t)r * N_DIM] = alpha * acc[ti][tj][r];
        }
    }
}

// ---------------- host launcher ----------------

extern "C" void kernel_launch(void* const* d_in, const int* in_sizes, int n_in,
                              void* d_out, int out_size, void* d_ws, size_t ws_size,
                              hipStream_t stream) {
    (void)in_sizes; (void)n_in; (void)out_size; (void)ws_size;

    const float* x     = (const float*)d_in[0];   // [M,K] f32
    const float* w     = (const float*)d_in[1];   // [N,K] f32
    const float* alpha = (const float*)d_in[2];   // [1]   f32
    float*       out   = (float*)d_out;           // [M,N] f32

    unsigned short* xb = (unsigned short*)d_ws;                               // 64 MB
    unsigned short* wb = (unsigned short*)((char*)d_ws + (size_t)M_DIM * K_DIM * 2);  // 8 MB

    {
        int n4 = (M_DIM * K_DIM) / 4;
        cvt_x_bf16<<<n4 / 256, 256, 0, stream>>>(x, xb, n4);
    }
    {
        int n4 = (N_DIM * K_DIM) / 4;
        cvt_w_sign<<<n4 / 256, 256, 0, stream>>>(w, wb, n4);
    }

    dim3 grid(N_DIM / BN, M_DIM / BM);   // (8, 128)
    bgemm_wmma<<<grid, 256, LDS_BYTES, stream>>>(xb, wb, alpha, out);
}
